// Density_loss_12378095747603
// MI455X (gfx1250) — compile-verified, matrix operand order checked by reading, not run on hardware
//
#include <hip/hip_runtime.h>
#include <hip/hip_bf16.h>
#include <math.h>

typedef __attribute__((ext_vector_type(2)))  float  v2f;
typedef __attribute__((ext_vector_type(8)))  float  v8f;
typedef __attribute__((ext_vector_type(8)))  __bf16 v8bf;
typedef __attribute__((ext_vector_type(16))) __bf16 v16bf;

#define N_ROWS 8192
#define DIM    256
#define KSTEPS 64          // 256 / 4   (f32 path)
#define NCHUNK 8           // 256 / 32  (bf16 path)
#define NTILES 512         // 8192 / 16
#define NWAVES 4
#define HINGE  0.01f
#define BIGF   3.0e38f

// ---------------------------------------------------------------------------
// Kernel 1: squared norms. One wave per row; rows [0,8192) = source -> sq[0..],
// rows [8192,16384) = target -> sq[8192..].
// ---------------------------------------------------------------------------
__global__ __launch_bounds__(256) void sqnorm_kernel(
    const float* __restrict__ src, const float* __restrict__ tgt,
    float* __restrict__ sq) {
  int wave = threadIdx.x >> 5;
  int lane = threadIdx.x & 31;
  int row  = blockIdx.x * 8 + wave;             // 0 .. 16383
  const float* base = (row < N_ROWS) ? (src + (size_t)row * DIM)
                                     : (tgt + (size_t)(row - N_ROWS) * DIM);
  const float4* b4 = (const float4*)base;       // 64 float4 per row
  float4 a = b4[lane];
  float4 b = b4[lane + 32];
  float s = a.x * a.x + a.y * a.y + a.z * a.z + a.w * a.w
          + b.x * b.x + b.y * b.y + b.z * b.z + b.w * b.w;
#pragma unroll
  for (int off = 16; off >= 1; off >>= 1) s += __shfl_xor(s, off);
  if (lane == 0) sq[row] = s;
}

// ---------------------------------------------------------------------------
// Kernel 1b: f32 -> (hi, lo) bf16 split planes, row-major.
// rows [0,8192) = source, [8192,16384) = target.
// ---------------------------------------------------------------------------
__global__ __launch_bounds__(256) void convert_kernel(
    const float* __restrict__ src, const float* __restrict__ tgt,
    __bf16* __restrict__ sHi, __bf16* __restrict__ sLo,
    __bf16* __restrict__ tHi, __bf16* __restrict__ tLo) {
  int row = blockIdx.x;            // 0 .. 16383
  int k   = threadIdx.x;           // 0 .. 255
  bool isT = row >= N_ROWS;
  int rr = isT ? row - N_ROWS : row;
  size_t idx = (size_t)rr * DIM + k;
  float x = (isT ? tgt : src)[idx];
  __bf16 hi = (__bf16)x;                       // RNE
  __bf16 lo = (__bf16)(x - (float)hi);         // residual
  (isT ? tHi : sHi)[idx] = hi;
  (isT ? tLo : sLo)[idx] = lo;
}

// Branchless insert of x into sorted 5-smallest list t0<=t1<=t2<=t3<=t4
__device__ __forceinline__ void ins5(float& t0, float& t1, float& t2,
                                     float& t3, float& t4, float x) {
  float m;
  m = fminf(t0, x); x = fmaxf(t0, x); t0 = m;
  m = fminf(t1, x); x = fmaxf(t1, x); t1 = m;
  m = fminf(t2, x); x = fmaxf(t2, x); t2 = m;
  m = fminf(t3, x); x = fmaxf(t3, x); t3 = m;
  t4 = fminf(t4, x);
}

// bf16 WMMA fragment: lane needs K runs [8h,8h+8) and [16+8h,24+8h) of the
// 32-wide chunk -> two 16B loads from a row-major bf16 plane.
__device__ __forceinline__ v16bf load_frag(const __bf16* p) {
  v8bf a = *(const v8bf*)p;
  v8bf b = *(const v8bf*)(p + 16);
  v16bf r;
#pragma unroll
  for (int i = 0; i < 8; ++i) { r[i] = a[i]; r[i + 8] = b[i]; }
  return r;
}

// ---------------------------------------------------------------------------
// Kernel 2 (BF16x3 path): fused gram-tile + per-row top-5.
// Block = 16-row strip, 4 waves split the 512 column tiles.
// dot = ah*bh + al*bh + ah*bl  (f32 accumulate) ~= fp32-exact.
// Key = |t_n|^2 - 2*dot (monotone in distance; sqrt deferred to final 5).
// ---------------------------------------------------------------------------
__global__ __launch_bounds__(32 * NWAVES) void topk_bf16_kernel(
    const __bf16* __restrict__ sHi, const __bf16* __restrict__ sLo,
    const __bf16* __restrict__ tHi, const __bf16* __restrict__ tLo,
    const float* __restrict__ sqs, const float* __restrict__ sqt,
    float* __restrict__ rowsum) {
  // padded stage: row stride 20 dwords -> 16B-aligned rows
  __shared__ __align__(16) float stage[NWAVES][16 * 20];
  __shared__ float mtop[NWAVES][2][16][5];

  const int w    = threadIdx.x >> 5;
  const int lane = threadIdx.x & 31;
  const int r    = lane & 15;
  const int h    = lane >> 4;
  const int ibase = blockIdx.x * 16;

  // Preload A strip fragments: 8 chunks x (hi,lo) x v16bf = 128 VGPRs.
  v16bf ah[NCHUNK], al[NCHUNK];
  {
    const __bf16* aph = sHi + (size_t)(ibase + r) * DIM + 8 * h;
    const __bf16* apl = sLo + (size_t)(ibase + r) * DIM + 8 * h;
#pragma unroll
    for (int c = 0; c < NCHUNK; ++c) {
      ah[c] = load_frag(aph + 32 * c);
      al[c] = load_frag(apl + 32 * c);
    }
  }

  float t0 = BIGF, t1 = BIGF, t2 = BIGF, t3 = BIGF, t4 = BIGF;

  for (int j = w; j < NTILES; j += NWAVES) {
    const int jbase = j * 16;
    v8f acc = {};
    const __bf16* bph = tHi + (size_t)(jbase + r) * DIM + 8 * h;
    const __bf16* bpl = tLo + (size_t)(jbase + r) * DIM + 8 * h;
#pragma unroll
    for (int c = 0; c < NCHUNK; ++c) {
      v16bf bh = load_frag(bph + 32 * c);
      v16bf bl = load_frag(bpl + 32 * c);
      acc = __builtin_amdgcn_wmma_f32_16x16x32_bf16(
          false, ah[c], false, bh, (short)0, acc, false, false);
      acc = __builtin_amdgcn_wmma_f32_16x16x32_bf16(
          false, al[c], false, bh, (short)0, acc, false, false);
      acc = __builtin_amdgcn_wmma_f32_16x16x32_bf16(
          false, ah[c], false, bl, (short)0, acc, false, false);
    }

    // Epilogue: key[m][n] = sqt[n] - 2*dot.  C layout: vgpr v -> M = v + 8*h,
    // N = r.  Stage through LDS so rows land contiguous.
    const float sqtn = sqt[jbase + r];
#pragma unroll
    for (int v = 0; v < 8; ++v) {
      stage[w][(v + 8 * h) * 20 + r] = fmaf(-2.0f, acc[v], sqtn);
    }
    __builtin_amdgcn_wave_barrier();
    asm volatile("s_wait_dscnt 0" ::: "memory");   // wave-local DS ordering
    __builtin_amdgcn_wave_barrier();

    // All 32 lanes scan: lane (r,h) scans row r, columns [8h, 8h+8).
    const float4* rowp = (const float4*)&stage[w][r * 20 + 8 * h];
    float4 q0 = rowp[0], q1 = rowp[1];
    ins5(t0, t1, t2, t3, t4, q0.x); ins5(t0, t1, t2, t3, t4, q0.y);
    ins5(t0, t1, t2, t3, t4, q0.z); ins5(t0, t1, t2, t3, t4, q0.w);
    ins5(t0, t1, t2, t3, t4, q1.x); ins5(t0, t1, t2, t3, t4, q1.y);
    ins5(t0, t1, t2, t3, t4, q1.z); ins5(t0, t1, t2, t3, t4, q1.w);
  }

  // 8 partial lists per row (4 waves x 2 half-lanes): merge, hinge, sum.
  mtop[w][h][r][0] = t0; mtop[w][h][r][1] = t1; mtop[w][h][r][2] = t2;
  mtop[w][h][r][3] = t3; mtop[w][h][r][4] = t4;
  __syncthreads();
  if (w == 0 && h == 0) {
    float u0 = BIGF, u1 = BIGF, u2 = BIGF, u3 = BIGF, u4 = BIGF;
#pragma unroll
    for (int ww = 0; ww < NWAVES; ++ww)
#pragma unroll
      for (int hh = 0; hh < 2; ++hh)
#pragma unroll
        for (int i = 0; i < 5; ++i)
          ins5(u0, u1, u2, u3, u4, mtop[ww][hh][r][i]);
    const float ss = sqs[ibase + r];
    float sum = 0.0f, d;
    d = sqrtf(fmaxf(u0 + ss, 0.0f)); sum += fmaxf(d - HINGE, 0.0f);
    d = sqrtf(fmaxf(u1 + ss, 0.0f)); sum += fmaxf(d - HINGE, 0.0f);
    d = sqrtf(fmaxf(u2 + ss, 0.0f)); sum += fmaxf(d - HINGE, 0.0f);
    d = sqrtf(fmaxf(u3 + ss, 0.0f)); sum += fmaxf(d - HINGE, 0.0f);
    d = sqrtf(fmaxf(u4 + ss, 0.0f)); sum += fmaxf(d - HINGE, 0.0f);
    rowsum[ibase + r] = sum;
  }
}

// ---------------------------------------------------------------------------
// Kernel 2 (fallback, exact f32 WMMA path) — used if ws can't hold bf16 planes.
// ---------------------------------------------------------------------------
__global__ __launch_bounds__(32 * NWAVES) void topk_f32_kernel(
    const float* __restrict__ src, const float* __restrict__ tgt,
    const float* __restrict__ sqs, const float* __restrict__ sqt,
    float* __restrict__ rowsum) {
  __shared__ __align__(16) float stage[NWAVES][16 * 20];
  __shared__ float mtop[NWAVES][2][16][5];

  const int w    = threadIdx.x >> 5;
  const int lane = threadIdx.x & 31;
  const int r    = lane & 15;
  const int h    = lane >> 4;
  const int ibase = blockIdx.x * 16;

  const float* ap = src + (size_t)(ibase + r) * DIM + 2 * h;
  v2f a[KSTEPS];
#pragma unroll
  for (int k = 0; k < KSTEPS; ++k) a[k] = *(const v2f*)(ap + 4 * k);

  float t0 = BIGF, t1 = BIGF, t2 = BIGF, t3 = BIGF, t4 = BIGF;

  for (int j = w; j < NTILES; j += NWAVES) {
    const int jbase = j * 16;
    v8f acc = {};
    const float* bp = tgt + (size_t)(jbase + r) * DIM + 2 * h;
#pragma unroll
    for (int k = 0; k < KSTEPS; ++k) {
      v2f b = *(const v2f*)(bp + 4 * k);
      acc = __builtin_amdgcn_wmma_f32_16x16x4_f32(
          false, a[k], false, b, (short)0, acc, false, false);
    }
    const float sqtn = sqt[jbase + r];
#pragma unroll
    for (int v = 0; v < 8; ++v) {
      stage[w][(v + 8 * h) * 20 + r] = fmaf(-2.0f, acc[v], sqtn);
    }
    __builtin_amdgcn_wave_barrier();
    asm volatile("s_wait_dscnt 0" ::: "memory");
    __builtin_amdgcn_wave_barrier();

    const float4* rowp = (const float4*)&stage[w][r * 20 + 8 * h];
    float4 q0 = rowp[0], q1 = rowp[1];
    ins5(t0, t1, t2, t3, t4, q0.x); ins5(t0, t1, t2, t3, t4, q0.y);
    ins5(t0, t1, t2, t3, t4, q0.z); ins5(t0, t1, t2, t3, t4, q0.w);
    ins5(t0, t1, t2, t3, t4, q1.x); ins5(t0, t1, t2, t3, t4, q1.y);
    ins5(t0, t1, t2, t3, t4, q1.z); ins5(t0, t1, t2, t3, t4, q1.w);
  }

  mtop[w][h][r][0] = t0; mtop[w][h][r][1] = t1; mtop[w][h][r][2] = t2;
  mtop[w][h][r][3] = t3; mtop[w][h][r][4] = t4;
  __syncthreads();
  if (w == 0 && h == 0) {
    float u0 = BIGF, u1 = BIGF, u2 = BIGF, u3 = BIGF, u4 = BIGF;
#pragma unroll
    for (int ww = 0; ww < NWAVES; ++ww)
#pragma unroll
      for (int hh = 0; hh < 2; ++hh)
#pragma unroll
        for (int i = 0; i < 5; ++i)
          ins5(u0, u1, u2, u3, u4, mtop[ww][hh][r][i]);
    const float ss = sqs[ibase + r];
    float sum = 0.0f, d;
    d = sqrtf(fmaxf(u0 + ss, 0.0f)); sum += fmaxf(d - HINGE, 0.0f);
    d = sqrtf(fmaxf(u1 + ss, 0.0f)); sum += fmaxf(d - HINGE, 0.0f);
    d = sqrtf(fmaxf(u2 + ss, 0.0f)); sum += fmaxf(d - HINGE, 0.0f);
    d = sqrtf(fmaxf(u3 + ss, 0.0f)); sum += fmaxf(d - HINGE, 0.0f);
    d = sqrtf(fmaxf(u4 + ss, 0.0f)); sum += fmaxf(d - HINGE, 0.0f);
    rowsum[ibase + r] = sum;
  }
}

// ---------------------------------------------------------------------------
// Kernel 3: deterministic final reduction of 8192 per-row sums -> mean.
// ---------------------------------------------------------------------------
__global__ __launch_bounds__(256) void reduce_kernel(
    const float* __restrict__ rowsum, float* __restrict__ out) {
  __shared__ float sm[256];
  float acc = 0.0f;
  for (int i = threadIdx.x; i < N_ROWS; i += 256) acc += rowsum[i];
  sm[threadIdx.x] = acc;
  __syncthreads();
#pragma unroll
  for (int s = 128; s > 0; s >>= 1) {
    if (threadIdx.x < s) sm[threadIdx.x] += sm[threadIdx.x + s];
    __syncthreads();
  }
  if (threadIdx.x == 0) out[0] = sm[0] * (1.0f / (float)(N_ROWS * 5));
}

extern "C" void kernel_launch(void* const* d_in, const int* in_sizes, int n_in,
                              void* d_out, int out_size, void* d_ws, size_t ws_size,
                              hipStream_t stream) {
  const float* src = (const float*)d_in[0];
  const float* tgt = (const float*)d_in[1];
  // d_in[2] = top_k (always 5 in this problem)
  float* ws     = (float*)d_ws;
  float* sqs    = ws;              // [8192]
  float* sqt    = ws + N_ROWS;     // [8192]
  float* rowsum = ws + 2 * N_ROWS; // [8192]

  const size_t HEAD  = 3 * (size_t)N_ROWS * sizeof(float);     // 96 KB
  const size_t PLANE = (size_t)N_ROWS * DIM;                   // elems per plane
  const size_t NEED  = HEAD + 4 * PLANE * sizeof(unsigned short); // +16 MB

  sqnorm_kernel<<<(2 * N_ROWS) / 8, 256, 0, stream>>>(src, tgt, ws);

  if (ws_size >= NEED) {
    __bf16* planes = (__bf16*)((char*)d_ws + HEAD);
    __bf16* sHi = planes;
    __bf16* sLo = planes + PLANE;
    __bf16* tHi = planes + 2 * PLANE;
    __bf16* tLo = planes + 3 * PLANE;
    convert_kernel<<<2 * N_ROWS, 256, 0, stream>>>(src, tgt, sHi, sLo, tHi, tLo);
    topk_bf16_kernel<<<NTILES, 32 * NWAVES, 0, stream>>>(
        sHi, sLo, tHi, tLo, sqs, sqt, rowsum);
  } else {
    topk_f32_kernel<<<NTILES, 32 * NWAVES, 0, stream>>>(
        src, tgt, sqs, sqt, rowsum);
  }
  reduce_kernel<<<1, 256, 0, stream>>>(rowsum, (float*)d_out);
}
